// EGNN_10471130267789
// MI455X (gfx1250) — matrix operand order repaired and checked by reading, not compiled, not run on hardware
//
#include <hip/hip_runtime.h>
#include <stdint.h>

// ---- problem constants (match reference) ----
#define HID     256
#define DIN     512
#define DSTATE  64
#define DCONV   4
#define DTRANK  16
#define NLAYERS 4
#define NNODES  2048
#define NEDGES  63488
#define NPER    32
#define NGRAPH  64
#define KEDGE   517   // 2*HID + 1 + 4
#define KNODE   523   // 2*HID + 11
#define KPAD17  544   // pad of 517/523 to multiple of 32
#define NDBC    144   // DT_RANK + 2*D_STATE

typedef __attribute__((ext_vector_type(16))) __bf16 v16bf;
typedef __attribute__((ext_vector_type(8)))  float  v8f;

union FragAB { v16bf v; uint32_t u[8]; };
union FragC  { v8f   v; float    f[8]; };

__device__ __forceinline__ uint32_t f2bf(float f) {
  union { float f; uint32_t u; } c; c.f = f;
  uint32_t u = c.u;
  u += 0x7FFFu + ((u >> 16) & 1u);   // round-to-nearest-even
  return u >> 16;
}
__device__ __forceinline__ uint32_t pack2(float lo, float hi) {
  return f2bf(lo) | (f2bf(hi) << 16);
}
__device__ __forceinline__ float silu_f(float x) { return x / (1.f + __expf(-x)); }

// ---- A fetchers (fused gather/concat; used once per tile to fill LDS) ----
struct FetchPlain {
  const float* A; int lda;
  __device__ float operator()(int row, int k) const { return A[row * lda + k]; }
};
struct FetchEdge {  // ein = [h[src], h[dst], radial, edge_attr], K=517 (pad 544)
  const float* h; const float* radial; const float* eattr; const int* src; const int* dst;
  __device__ float operator()(int e, int k) const {
    if (k < HID)       return h[src[e] * HID + k];
    if (k < 2 * HID)   return h[dst[e] * HID + (k - HID)];
    if (k == 2 * HID)  return radial[e];
    if (k < KEDGE)     return eattr[e * 4 + (k - 2 * HID - 1)];
    return 0.f;
  }
};
struct FetchNode {  // hn = [mi, h, h0], K=523 (pad 544)
  const float* mi; const float* h; const float* h0;
  __device__ float operator()(int n, int k) const {
    if (k < HID)     return mi[n * HID + k];
    if (k < 2 * HID) return h[n * HID + (k - HID)];
    if (k < KNODE)   return h0[n * 11 + (k - 2 * HID)];
    return 0.f;
  }
};

// ---- weight repack: fp32 [K,N] -> bf16 B-fragment order [kstep][ntile][lane][8xu32],
// zero-padded to Kpad. B 32x16 in this file's contiguous k order (A matches): lanes0-15 K=0..15, lanes16-31 K=16..31.
__global__ void repack_w(const float* __restrict__ src, uint32_t* __restrict__ dst,
                         int K, int N, int Kpad) {
  int idx = blockIdx.x * blockDim.x + threadIdx.x;
  int tiles_n = N >> 4;
  int total = (Kpad >> 5) * tiles_n * 256;
  if (idx >= total) return;
  int v    = idx & 7;
  int lane = (idx >> 3) & 31;
  int rest = idx >> 8;
  int nt   = rest % tiles_n;
  int ks   = rest / tiles_n;
  int col  = nt * 16 + (lane & 15);
  int k0   = ks * 32 + ((lane >> 4) << 4) + 2 * v;
  float lo = (k0     < K) ? src[k0 * N + col]       : 0.f;
  float hi = (k0 + 1 < K) ? src[(k0 + 1) * N + col] : 0.f;
  dst[idx] = pack2(lo, hi);
}

// ---- staged WMMA GEMM: one block = one 16-row M-tile, 8 waves over N-tiles.
// A tile staged to LDS as bf16 once; inner loop = 2x ds_read_b128 + 2x global_load_b128 + wmma.
template <typename F>
__global__ void gemm_wmma_staged(F fetch, const uint32_t* __restrict__ Wswz,
                                 const float* __restrict__ bias, float* __restrict__ C,
                                 int N, int Kpad, int act) {
  extern __shared__ char smem_raw[];
  uint16_t* lds_a = (uint16_t*)smem_raw;
  const int ldk = Kpad + 8;                    // pad 16B to spread LDS banks
  const int m0  = blockIdx.x << 4;
  const int tid = threadIdx.x;

  for (int i = tid; i < 16 * Kpad; i += 256) {
    int row = i / Kpad, k = i - row * Kpad;
    lds_a[row * ldk + k] = (uint16_t)f2bf(fetch(m0 + row, k));
  }
  __syncthreads();

  const int lane = tid & 31;
  const int wid  = tid >> 5;
  const int half = lane >> 4;
  const int rrow = lane & 15;
  const int tiles_n = N >> 4;
  const int ksteps  = Kpad >> 5;
  const uint16_t* arow = lds_a + rrow * ldk + half * 16;

  for (int nt = wid; nt < tiles_n; nt += 8) {
    FragC acc;
#pragma unroll
    for (int i = 0; i < 8; ++i) acc.f[i] = 0.f;
    const uint4* wq = (const uint4*)Wswz + ((size_t)nt * 32 + lane) * 2;
    const size_t wstride = (size_t)tiles_n * 64;   // uint4 per k-step

    for (int ks = 0; ks < ksteps; ++ks) {
      const uint16_t* ap = arow + (ks << 5);
      uint4 a0 = *(const uint4*)ap;              // K: half*16 + 0..7   (pairs v0..3)
      uint4 a1 = *(const uint4*)(ap + 8);        // K: half*16 + 8..15 (pairs v4..7)
      uint4 b0 = wq[0];
      uint4 b1 = wq[1];
      __builtin_prefetch((const void*)(wq + wstride), 0, 1);
      wq += wstride;
      FragAB a, b;
      a.u[0] = a0.x; a.u[1] = a0.y; a.u[2] = a0.z; a.u[3] = a0.w;
      a.u[4] = a1.x; a.u[5] = a1.y; a.u[6] = a1.z; a.u[7] = a1.w;
      b.u[0] = b0.x; b.u[1] = b0.y; b.u[2] = b0.z; b.u[3] = b0.w;
      b.u[4] = b1.x; b.u[5] = b1.y; b.u[6] = b1.z; b.u[7] = b1.w;
      acc.v = __builtin_amdgcn_wmma_f32_16x16x32_bf16(false, a.v, false, b.v,
                                                      (short)0, acc.v, false, false);
    }
    const int col = nt * 16 + rrow;
    const float bv = bias ? bias[col] : 0.f;
#pragma unroll
    for (int r = 0; r < 8; ++r) {
      float v = acc.f[r] + bv;
      if (act) v = silu_f(v);
      C[(m0 + r + half * 8) * N + col] = v;
    }
  }
}

__global__ void embed_kernel(const float* __restrict__ h0, const float* __restrict__ w,
                             const float* __restrict__ b, float* __restrict__ h) {
  int idx = blockIdx.x * blockDim.x + threadIdx.x;
  if (idx >= NNODES * HID) return;
  int n = idx >> 8, j = idx & 255;
  float acc = b[j];
#pragma unroll
  for (int k = 0; k < 11; ++k) acc += h0[n * 11 + k] * w[k * HID + j];
  h[idx] = acc;
}

__global__ void radial_kernel(const float* __restrict__ x, const int* __restrict__ src,
                              const int* __restrict__ dst, float* __restrict__ radial) {
  int e = blockIdx.x * blockDim.x + threadIdx.x;
  if (e >= NEDGES) return;
  int s = src[e], d = dst[e];
  float dx = x[s * 3 + 0] - x[d * 3 + 0];
  float dy = x[s * 3 + 1] - x[d * 3 + 1];
  float dz = x[s * 3 + 2] - x[d * 3 + 2];
  radial[e] = dx * dx + dy * dy + dz * dz;
}

// edges are generated sorted by src: node n owns edges [n*31, n*31+31) -> no atomics
__global__ void segsum_kernel(const float* __restrict__ ef, const float* __restrict__ emask,
                              float* __restrict__ mi) {
  int idx = blockIdx.x * blockDim.x + threadIdx.x;
  if (idx >= NNODES * HID) return;
  int n = idx >> 8, j = idx & 255;
  float acc = 0.f;
  int e0 = n * (NPER - 1);
  for (int i = 0; i < NPER - 1; ++i) acc += ef[(e0 + i) * HID + j] * emask[e0 + i];
  mi[idx] = acc;
}

__global__ void ln_clip_kernel(const float* __restrict__ in, const float* __restrict__ g,
                               const float* __restrict__ b, float* __restrict__ out) {
  __shared__ float red[HID];
  int n = blockIdx.x, j = threadIdx.x;
  float v = in[n * HID + j];
  red[j] = v; __syncthreads();
  for (int s = 128; s > 0; s >>= 1) { if (j < s) red[j] += red[j + s]; __syncthreads(); }
  float mu = red[0] * (1.f / HID); __syncthreads();
  float d = v - mu;
  red[j] = d * d; __syncthreads();
  for (int s = 128; s > 0; s >>= 1) { if (j < s) red[j] += red[j + s]; __syncthreads(); }
  float var = red[0] * (1.f / HID);
  float y = d * rsqrtf(var + 1e-5f) * g[j] + b[j];
  out[n * HID + j] = fminf(10.f, fmaxf(-10.f, y));
}

__global__ void conv_silu_kernel(const float* __restrict__ xz, const float* __restrict__ cw,
                                 const float* __restrict__ cb, float* __restrict__ u) {
  int idx = blockIdx.x * blockDim.x + threadIdx.x;
  if (idx >= NNODES * DIN) return;
  int t = idx / DIN, d = idx % DIN;
  float acc = cb[d];
#pragma unroll
  for (int k = 0; k < DCONV; ++k) {
    int tt = t + k - (DCONV - 1);
    if (tt >= 0) acc += xz[tt * (2 * DIN) + d] * cw[d * DCONV + k];
  }
  u[idx] = silu_f(acc);
}

__global__ void dt_kernel(const float* __restrict__ dbc, const float* __restrict__ dtw,
                          const float* __restrict__ dtb, float* __restrict__ dt) {
  int idx = blockIdx.x * blockDim.x + threadIdx.x;
  if (idx >= NNODES * DIN) return;
  int t = idx / DIN, d = idx % DIN;
  float acc = dtb[d];
#pragma unroll
  for (int r = 0; r < DTRANK; ++r) acc += dbc[t * NDBC + r] * dtw[r * DIN + d];
  dt[idx] = (acc > 20.f) ? acc : log1pf(__expf(acc));   // softplus
}

// selective scan: one lane per channel d; state[64] + A-row[64] live in VGPRs
__global__ void scan_kernel(const float* __restrict__ dt, const float* __restrict__ dbc,
                            const float* __restrict__ u, const float* __restrict__ xz,
                            const float* __restrict__ A_log, const float* __restrict__ Dp,
                            float* __restrict__ ym) {
  int d = blockIdx.x * 32 + threadIdx.x;
  if (d >= DIN) return;
  float a[DSTATE], s[DSTATE];
#pragma unroll
  for (int j = 0; j < DSTATE; ++j) { a[j] = -__expf(A_log[d * DSTATE + j]); s[j] = 0.f; }
  const float Dv = Dp[d];
  for (int t = 0; t < NNODES; ++t) {
    const float dtv = dt[t * DIN + d];
    const float uv  = u[t * DIN + d];
    const float du  = dtv * uv;
    const float* Bp = dbc + t * NDBC + DTRANK;
    const float* Cp = Bp + DSTATE;
    float acc = 0.f;
#pragma unroll
    for (int j = 0; j < DSTATE; ++j) {
      s[j] = __expf(dtv * a[j]) * s[j] + du * Bp[j];
      acc += s[j] * Cp[j];
    }
    const float y = acc + Dv * uv;
    const float z = xz[t * (2 * DIN) + DIN + d];
    ym[t * DIN + d] = y * silu_f(z);
  }
}

__global__ void pool_kernel(const float* __restrict__ d2, const float* __restrict__ nmask,
                            float* __restrict__ hp) {
  int idx = blockIdx.x * blockDim.x + threadIdx.x;
  if (idx >= NGRAPH * HID) return;
  int g = idx >> 8, j = idx & 255;
  float acc = 0.f;
  for (int i = 0; i < NPER; ++i) acc += d2[(g * NPER + i) * HID + j] * nmask[g * NPER + i];
  hp[idx] = acc;
}

__global__ void head_kernel(const float* __restrict__ g1, const float* __restrict__ w2,
                            const float* __restrict__ b2, float* __restrict__ out) {
  int g = blockIdx.x * blockDim.x + threadIdx.x;
  if (g >= NGRAPH) return;
  float acc = b2[0];
  for (int j = 0; j < HID; ++j) acc += g1[g * HID + j] * w2[j];
  out[g] = acc;
}

// ---- layer parameter indices in flattened d_in ----
enum { P_E_W1 = 0, P_E_B1, P_E_W2, P_E_B2, P_N_W1, P_N_B1, P_N_W2, P_N_B2,
       P_LN_G, P_LN_B, P_IN_PROJ, P_CONV_W, P_CONV_B, P_X_PROJ, P_DT_W, P_DT_B,
       P_A_LOG, P_DP, P_OUT_PROJ, LPN };

extern "C" void kernel_launch(void* const* d_in, const int* in_sizes, int n_in,
                              void* d_out, int out_size, void* d_ws, size_t ws_size,
                              hipStream_t stream) {
  (void)in_sizes; (void)n_in; (void)out_size; (void)ws_size;
  const float* h0        = (const float*)d_in[0];
  const float* x         = (const float*)d_in[1];
  const int*   edges     = (const int*)d_in[2];
  const float* edge_attr = (const float*)d_in[3];
  const float* node_mask = (const float*)d_in[4];
  const float* edge_mask = (const float*)d_in[5];
  // d_in[6] = n_nodes scalar (compile-time NPER)
  const float* emb_w = (const float*)d_in[7];
  const float* emb_b = (const float*)d_in[8];
  const int LPB = 9;
  auto LP = [&](int l, int p) { return (const float*)d_in[LPB + l * LPN + p]; };
  const float* dec_w1 = (const float*)d_in[LPB + NLAYERS * LPN + 0];
  const float* dec_b1 = (const float*)d_in[LPB + NLAYERS * LPN + 1];
  const float* dec_w2 = (const float*)d_in[LPB + NLAYERS * LPN + 2];
  const float* dec_b2 = (const float*)d_in[LPB + NLAYERS * LPN + 3];
  const float* g_w1   = (const float*)d_in[LPB + NLAYERS * LPN + 4];
  const float* g_b1   = (const float*)d_in[LPB + NLAYERS * LPN + 5];
  const float* g_w2   = (const float*)d_in[LPB + NLAYERS * LPN + 6];
  const float* g_b2   = (const float*)d_in[LPB + NLAYERS * LPN + 7];

  const int* src = edges;
  const int* dst = edges + NEDGES;

  // ---- workspace carve-out ----
  char* ws = (char*)d_ws;
  size_t off = 0;
  auto alloc = [&](size_t bytes) -> void* {
    void* p = ws + off; off += (bytes + 255) & ~(size_t)255; return p;
  };
  float* h    = (float*)alloc((size_t)NNODES * HID * 4);
  float* rad  = (float*)alloc((size_t)NEDGES * 4);
  float* ef1  = (float*)alloc((size_t)NEDGES * HID * 4);
  float* ef2  = (float*)alloc((size_t)NEDGES * HID * 4);
  float* mi   = (float*)alloc((size_t)NNODES * HID * 4);
  float* hn1  = (float*)alloc((size_t)NNODES * HID * 4);
  float* hn2  = (float*)alloc((size_t)NNODES * HID * 4);
  float* hln  = (float*)alloc((size_t)NNODES * HID * 4);
  float* xz   = (float*)alloc((size_t)NNODES * 2 * DIN * 4);
  float* ubuf = (float*)alloc((size_t)NNODES * DIN * 4);
  float* dbc  = (float*)alloc((size_t)NNODES * NDBC * 4);
  float* dtv  = (float*)alloc((size_t)NNODES * DIN * 4);
  float* ym   = (float*)alloc((size_t)NNODES * DIN * 4);
  float* dd1  = (float*)alloc((size_t)NNODES * HID * 4);
  float* dd2  = (float*)alloc((size_t)NNODES * HID * 4);
  float* hp   = (float*)alloc((size_t)NGRAPH * HID * 4);
  float* g1   = (float*)alloc((size_t)NGRAPH * HID * 4);

  auto repack = [&](const float* srcw, int K, int N, int Kpad) -> uint32_t* {
    uint32_t* dstw = (uint32_t*)alloc((size_t)Kpad * N * 2);
    int total = (Kpad >> 5) * (N >> 4) * 256;
    repack_w<<<dim3((total + 255) / 256), dim3(256), 0, stream>>>(srcw, dstw, K, N, Kpad);
    return dstw;
  };

  uint32_t *we1[NLAYERS], *we2[NLAYERS], *wn1[NLAYERS], *wn2[NLAYERS];
  uint32_t *win[NLAYERS], *wx[NLAYERS], *wout[NLAYERS];
  for (int l = 0; l < NLAYERS; ++l) {
    we1[l]  = repack(LP(l, P_E_W1),     KEDGE, HID,     KPAD17);
    we2[l]  = repack(LP(l, P_E_W2),     HID,   HID,     HID);
    wn1[l]  = repack(LP(l, P_N_W1),     KNODE, HID,     KPAD17);
    wn2[l]  = repack(LP(l, P_N_W2),     HID,   HID,     HID);
    win[l]  = repack(LP(l, P_IN_PROJ),  HID,   2 * DIN, HID);
    wx[l]   = repack(LP(l, P_X_PROJ),   DIN,   NDBC,    DIN);
    wout[l] = repack(LP(l, P_OUT_PROJ), DIN,   HID,     DIN);
  }
  uint32_t* wdec1 = repack(dec_w1, HID, HID, HID);
  uint32_t* wdec2 = repack(dec_w2, HID, HID, HID);
  uint32_t* wg1   = repack(g_w1,   HID, HID, HID);

  auto shmem = [](int Kpad) { return (size_t)16 * (Kpad + 8) * 2; };

  embed_kernel<<<dim3(NNODES * HID / 256), dim3(256), 0, stream>>>(h0, emb_w, emb_b, h);
  radial_kernel<<<dim3((NEDGES + 255) / 256), dim3(256), 0, stream>>>(x, src, dst, rad);

  for (int l = 0; l < NLAYERS; ++l) {
    FetchEdge fe{h, rad, edge_attr, src, dst};
    gemm_wmma_staged<FetchEdge><<<dim3(NEDGES >> 4), dim3(256), shmem(KPAD17), stream>>>(
        fe, we1[l], LP(l, P_E_B1), ef1, HID, KPAD17, 1);
    gemm_wmma_staged<FetchPlain><<<dim3(NEDGES >> 4), dim3(256), shmem(HID), stream>>>(
        FetchPlain{ef1, HID}, we2[l], LP(l, P_E_B2), ef2, HID, HID, 1);
    segsum_kernel<<<dim3(NNODES * HID / 256), dim3(256), 0, stream>>>(ef2, edge_mask, mi);

    FetchNode fn{mi, h, h0};
    gemm_wmma_staged<FetchNode><<<dim3(NNODES >> 4), dim3(256), shmem(KPAD17), stream>>>(
        fn, wn1[l], LP(l, P_N_B1), hn1, HID, KPAD17, 1);
    gemm_wmma_staged<FetchPlain><<<dim3(NNODES >> 4), dim3(256), shmem(HID), stream>>>(
        FetchPlain{hn1, HID}, wn2[l], LP(l, P_N_B2), hn2, HID, HID, 0);
    ln_clip_kernel<<<dim3(NNODES), dim3(HID), 0, stream>>>(hn2, LP(l, P_LN_G), LP(l, P_LN_B), hln);

    // Mamba block
    gemm_wmma_staged<FetchPlain><<<dim3(NNODES >> 4), dim3(256), shmem(HID), stream>>>(
        FetchPlain{hln, HID}, win[l], (const float*)nullptr, xz, 2 * DIN, HID, 0);
    conv_silu_kernel<<<dim3(NNODES * DIN / 256), dim3(256), 0, stream>>>(
        xz, LP(l, P_CONV_W), LP(l, P_CONV_B), ubuf);
    gemm_wmma_staged<FetchPlain><<<dim3(NNODES >> 4), dim3(256), shmem(DIN), stream>>>(
        FetchPlain{ubuf, DIN}, wx[l], (const float*)nullptr, dbc, NDBC, DIN, 0);
    dt_kernel<<<dim3(NNODES * DIN / 256), dim3(256), 0, stream>>>(
        dbc, LP(l, P_DT_W), LP(l, P_DT_B), dtv);
    scan_kernel<<<dim3(DIN / 32), dim3(32), 0, stream>>>(
        dtv, dbc, ubuf, xz, LP(l, P_A_LOG), LP(l, P_DP), ym);
    gemm_wmma_staged<FetchPlain><<<dim3(NNODES >> 4), dim3(256), shmem(DIN), stream>>>(
        FetchPlain{ym, DIN}, wout[l], (const float*)nullptr, h, HID, DIN, 0);
  }

  gemm_wmma_staged<FetchPlain><<<dim3(NNODES >> 4), dim3(256), shmem(HID), stream>>>(
      FetchPlain{h, HID}, wdec1, dec_b1, dd1, HID, HID, 1);
  gemm_wmma_staged<FetchPlain><<<dim3(NNODES >> 4), dim3(256), shmem(HID), stream>>>(
      FetchPlain{dd1, HID}, wdec2, dec_b2, dd2, HID, HID, 0);
  pool_kernel<<<dim3(NGRAPH * HID / 256), dim3(256), 0, stream>>>(dd2, node_mask, hp);
  gemm_wmma_staged<FetchPlain><<<dim3(NGRAPH >> 4), dim3(256), shmem(HID), stream>>>(
      FetchPlain{hp, HID}, wg1, g_b1, g1, HID, HID, 1);
  head_kernel<<<dim3(1), dim3(NGRAPH), 0, stream>>>(g1, g_w2, g_b2, (float*)d_out);
}
